// PruningAgent_52690658787736
// MI455X (gfx1250) — matrix operand order repaired and checked by reading, not compiled, not run on hardware
//
#include <hip/hip_runtime.h>
#include <hip/hip_bf16.h>
#include <math.h>

// ---------------------------------------------------------------------------
// Problem constants (match reference)
// ---------------------------------------------------------------------------
#define H_DIM    128
#define F_DIM    256
#define N_LAYER  40000
#define E_LAYER  640000
#define L_HEADS  8
#define NH_NODES 8192          // 2^13
#define EH_EDGES 131072        // 2^17
#define NT_NODES (N_LAYER + L_HEADS * NH_NODES)   // 105536 (multiple of 32)

typedef __attribute__((ext_vector_type(16))) __bf16          v16bf;
typedef __attribute__((ext_vector_type(8)))  float           v8f;
typedef __attribute__((ext_vector_type(8)))  unsigned short  us8v;

static __device__ __forceinline__ unsigned short f2bfu(float f) {
  unsigned int u = __float_as_uint(f);
  u += 0x7fffu + ((u >> 16) & 1u);           // round-to-nearest-even
  return (unsigned short)(u >> 16);
}

// ---------------------------------------------------------------------------
// Weight pre-shuffle: W[K][128] (f32) -> per-lane WMMA B fragments (bf16).
// Fragment (kc, ntile, lane) = 16 contiguous bf16:
//   element j = W[kc*32 + (lane>>4)*16 + j][ntile*16 + (lane&15)]
// so the GEMM loads each B fragment as one 32-byte vector (2x b128).
// ---------------------------------------------------------------------------
__global__ void k_shuffle_w(const float* __restrict__ W,
                            unsigned short* __restrict__ out, int K) {
  int t = blockIdx.x * blockDim.x + threadIdx.x;   // (K/32)*8*32 threads
  int total = (K >> 5) * 8 * 32;
  if (t >= total) return;
  int lane  = t & 31;
  int ntile = (t >> 5) & 7;
  int kc    = t >> 8;
  int half  = lane >> 4;
  int col   = ntile * 16 + (lane & 15);
  unsigned short* o = out + (size_t)t * 16;
#pragma unroll
  for (int j = 0; j < 16; ++j)
    o[j] = f2bfu(W[(size_t)(kc * 32 + half * 16 + j) * H_DIM + col]);
}

// ---------------------------------------------------------------------------
// WMMA GEMM:  Out[M,128] = act(A[M,K] @ W[K,128] + bias)
//   - 256 threads = 8 waves; block computes a 32x128 tile
//   - wave w owns columns [16w,16w+16); two 16x16 row tiles share one
//     B fragment per 32-wide K chunk (2 wmma per chunk)
//   - A staged to LDS as bf16 (converted from f32 when A_IS_F32)
//   - B fragments pre-shuffled (k_shuffle_w): one 32B vector load per chunk
//   - A/B use contiguous k order (k = 16*half + s), not ISA table order (wave32)
// ---------------------------------------------------------------------------
template<bool A_IS_F32, bool BIAS_RELU, bool OUT_BF16>
__global__ void gemm128_wmma(const void* __restrict__ Ain,
                             const unsigned short* __restrict__ Wb,
                             const float* __restrict__ bias,
                             void* __restrict__ Out, int K) {
  __shared__ __align__(16) unsigned short sA[32][32];

  const int t    = threadIdx.x;
  const int wave = t >> 5;
  const int lane = t & 31;
  const int half = lane >> 4;       // 0: lanes 0-15, 1: lanes 16-31
  const int l15  = lane & 15;
  const int row0 = blockIdx.x * 32;

  // staging assignment: 32 rows x 32 cols bf16, 4 elems (8 B) per thread
  const int ldr = t >> 3;           // 0..31
  const int ldc = (t & 7) * 4;      // 0,4,...,28

  const v16bf* __restrict__ Wv = (const v16bf*)Wb;

  v8f c0 = {};
  v8f c1 = {};

  for (int k0 = 0, kc = 0; k0 < K; k0 += 32, ++kc) {
    // ---- stage A tile (32x32 bf16) into LDS, vectorized ----
    if (A_IS_F32) {
      const float* p = (const float*)Ain + (size_t)(row0 + ldr) * K + k0 + ldc;
      if (k0 + 32 < K) __builtin_prefetch(p + 32, 0, 1);   // global_prefetch_b8
      float4 f = *(const float4*)p;
      unsigned int lo = (unsigned int)f2bfu(f.x) | ((unsigned int)f2bfu(f.y) << 16);
      unsigned int hi = (unsigned int)f2bfu(f.z) | ((unsigned int)f2bfu(f.w) << 16);
      *(uint2*)&sA[ldr][ldc] = make_uint2(lo, hi);
    } else {
      const unsigned short* p =
          (const unsigned short*)Ain + (size_t)(row0 + ldr) * K + k0 + ldc;
      if (k0 + 32 < K) __builtin_prefetch(p + 32, 0, 1);
      *(uint2*)&sA[ldr][ldc] = *(const uint2*)p;
    }
    __syncthreads();

    // ---- B fragment: one 32-byte contiguous vector load per lane ----
    v16bf b = Wv[(size_t)(kc * 8 + wave) * 32 + lane];

    // ---- A fragments: 2x 16-byte LDS vector loads each ----
    union Frag { us8v h[2]; v16bf v; };
    Frag a0, a1;
    a0.h[0] = *(const us8v*)&sA[l15][half * 16];
    a0.h[1] = *(const us8v*)&sA[l15][half * 16 + 8];
    a1.h[0] = *(const us8v*)&sA[16 + l15][half * 16];
    a1.h[1] = *(const us8v*)&sA[16 + l15][half * 16 + 8];

    c0 = __builtin_amdgcn_wmma_f32_16x16x32_bf16(false, a0.v, false, b,
                                                 (short)0, c0, false, false);
    c1 = __builtin_amdgcn_wmma_f32_16x16x32_bf16(false, a1.v, false, b,
                                                 (short)0, c1, false, false);
    __syncthreads();
  }

  // C/D layout: VGPR v -> M = v + 8*half, lane l15 -> N
  const int n = wave * 16 + l15;
  const float bval = BIAS_RELU ? bias[n] : 0.0f;
#pragma unroll
  for (int v = 0; v < 8; ++v) {
    int m0 = row0 + half * 8 + v;
    int m1 = m0 + 16;
    float x0 = c0[v], x1 = c1[v];
    if (BIAS_RELU) {
      x0 = fmaxf(x0 + bval, 0.0f);
      x1 = fmaxf(x1 + bval, 0.0f);
    }
    if (OUT_BF16) {
      ((unsigned short*)Out)[(size_t)m0 * H_DIM + n] = f2bfu(x0);
      ((unsigned short*)Out)[(size_t)m1 * H_DIM + n] = f2bfu(x1);
    } else {
      ((float*)Out)[(size_t)m0 * H_DIM + n] = x0;
      ((float*)Out)[(size_t)m1 * H_DIM + n] = x1;
    }
  }
}

// ---------------------------------------------------------------------------
// Small helper kernels
// ---------------------------------------------------------------------------
__global__ void k_fill1(float* p, int n) {
  int i = blockIdx.x * blockDim.x + threadIdx.x;
  if (i < n) p[i] = 1.0f;   // self-loop contribution to degree
}

__global__ void k_deg_layer(const int* __restrict__ ei, float* deg) {
  int e = blockIdx.x * blockDim.x + threadIdx.x;
  if (e < E_LAYER) atomicAdd(&deg[ei[E_LAYER + e]], 1.0f);
}

__global__ void k_deg_head(const int* __restrict__ hei, float* deg) {
  int t = blockIdx.x * blockDim.x + threadIdx.x;
  if (t >= L_HEADS * EH_EDGES) return;
  int g = t >> 17;                 // / EH_EDGES
  int e = t & (EH_EDGES - 1);
  int d = hei[((size_t)(g * 2 + 1)) * EH_EDGES + e];
  atomicAdd(&deg[N_LAYER + g * NH_NODES + d], 1.0f);
}

__global__ void k_finalize_dinv(float* p, int n) {
  int i = blockIdx.x * blockDim.x + threadIdx.x;
  if (i < n) p[i] = rsqrtf(p[i]);  // deg >= 1 always (self loop)
}

// G[i,:] = T[i,:] * dinv[i]^2   (self-loop term, also zero-initializes G)
__global__ void k_self_loop(const float* __restrict__ T,
                            const float* __restrict__ dinv,
                            float* __restrict__ G, int n) {
  int i = blockIdx.x * blockDim.x + threadIdx.x;
  if (i >= n) return;
  float di = dinv[i >> 7];
  G[i] = T[i] * di * di;
}

// one wave per edge; lane handles 4 features (float4 gather + 4 f32 atomics)
__global__ void k_edge_agg_layer(const int* __restrict__ ei,
                                 const float* __restrict__ T,
                                 const float* __restrict__ dinv,
                                 float* __restrict__ G) {
  long long tid = (long long)blockIdx.x * blockDim.x + threadIdx.x;
  int e = (int)(tid >> 5);
  int lane = (int)(tid & 31);
  if (e >= E_LAYER) return;
  int s = ei[e];
  int d = ei[E_LAYER + e];
  float sc = dinv[s] * dinv[d];
  float4 v = ((const float4*)(T + (size_t)s * H_DIM))[lane];
  float* g = G + (size_t)d * H_DIM + lane * 4;
  atomicAdd(g + 0, v.x * sc);
  atomicAdd(g + 1, v.y * sc);
  atomicAdd(g + 2, v.z * sc);
  atomicAdd(g + 3, v.w * sc);
}

__global__ void k_edge_agg_head(const int* __restrict__ hei,
                                const float* __restrict__ T,
                                const float* __restrict__ dinv,
                                float* __restrict__ G) {
  long long tid = (long long)blockIdx.x * blockDim.x + threadIdx.x;
  int ea = (int)(tid >> 5);
  int lane = (int)(tid & 31);
  if (ea >= L_HEADS * EH_EDGES) return;
  int g = ea >> 17;
  int e = ea & (EH_EDGES - 1);
  int base = N_LAYER + g * NH_NODES;
  int s = hei[((size_t)(g * 2 + 0)) * EH_EDGES + e] + base;
  int d = hei[((size_t)(g * 2 + 1)) * EH_EDGES + e] + base;
  float sc = dinv[s] * dinv[d];
  float4 v = ((const float4*)(T + (size_t)s * H_DIM))[lane];
  float* go = G + (size_t)d * H_DIM + lane * 4;
  atomicAdd(go + 0, v.x * sc);
  atomicAdd(go + 1, v.y * sc);
  atomicAdd(go + 2, v.z * sc);
  atomicAdd(go + 3, v.w * sc);
}

template<bool WRITE_F32>
__global__ void k_bias_relu_cvt(float* __restrict__ G,
                                const float* __restrict__ bias,
                                unsigned short* __restrict__ obf, int n) {
  int i = blockIdx.x * blockDim.x + threadIdx.x;
  if (i >= n) return;
  float v = fmaxf(G[i] + bias[i & (H_DIM - 1)], 0.0f);
  obf[i] = f2bfu(v);
  if (WRITE_F32) G[i] = v;
}

// sigmoid( U[node,:] . w2 + b2 ) — one wave per node, shfl reduction
__global__ void k_policy_out(const float* __restrict__ U,
                             const float* __restrict__ w2,
                             const float* __restrict__ b2,
                             float* __restrict__ out, int nrows) {
  long long tid = (long long)blockIdx.x * blockDim.x + threadIdx.x;
  int node = (int)(tid >> 5);
  int lane = (int)(tid & 31);
  if (node >= nrows) return;
  float4 u = ((const float4*)(U + (size_t)node * H_DIM))[lane];
  float4 w = ((const float4*)w2)[lane];
  float s = u.x * w.x + u.y * w.y + u.z * w.z + u.w * w.w;
#pragma unroll
  for (int o = 16; o > 0; o >>= 1) s += __shfl_xor(s, o, 32);
  if (lane == 0) out[node] = 1.0f / (1.0f + __expf(-(s + b2[0])));
}

__global__ void k_mask(float* out) {
  int i = threadIdx.x;
  if (i < L_HEADS)
    out[N_LAYER + L_HEADS * NH_NODES + i] = (out[i] > 0.5f) ? 1.0f : 0.0f;
}

__global__ void k_zero(float* p, int n) {
  int i = blockIdx.x * blockDim.x + threadIdx.x;
  if (i < n) p[i] = 0.0f;
}

// column sums over 128-row chunks of G's first N_LAYER rows -> acc[128]
__global__ void k_row_sum(const float* __restrict__ G, float* __restrict__ acc) {
  int f = threadIdx.x;                 // 128 threads
  int r0 = blockIdx.x * 128;
  float s = 0.0f;
  for (int r = 0; r < 128; ++r) {
    int row = r0 + r;
    if (row < N_LAYER) s += G[(size_t)row * H_DIM + f];
  }
  atomicAdd(&acc[f], s);
}

__global__ void k_value(const float* __restrict__ acc,
                        const float* __restrict__ w1, const float* __restrict__ b1,
                        const float* __restrict__ w2, const float* __restrict__ b2,
                        float* __restrict__ out) {
  __shared__ float m[128];
  __shared__ float red[128];
  int t = threadIdx.x;
  m[t] = acc[t] * (1.0f / (float)N_LAYER);
  __syncthreads();
  float s = b1[t];
  for (int k = 0; k < H_DIM; ++k) s += m[k] * w1[k * H_DIM + t];
  red[t] = fmaxf(s, 0.0f) * w2[t];
  __syncthreads();
  for (int o = 64; o > 0; o >>= 1) {
    if (t < o) red[t] += red[t + o];
    __syncthreads();
  }
  if (t == 0) out[0] = red[0] + b2[0];
}

// ---------------------------------------------------------------------------
// Host orchestration
// ---------------------------------------------------------------------------
static inline int cdiv(long long a, long long b) { return (int)((a + b - 1) / b); }

extern "C" void kernel_launch(void* const* d_in, const int* in_sizes, int n_in,
                              void* d_out, int out_size, void* d_ws, size_t ws_size,
                              hipStream_t stream) {
  (void)in_sizes; (void)n_in; (void)out_size; (void)ws_size;

  const float* x      = (const float*)d_in[0];
  const int*   ei     = (const int*)d_in[1];
  const float* head_x = (const float*)d_in[2];
  const int*   hei    = (const int*)d_in[3];
  const float* enc_w1 = (const float*)d_in[4];  const float* enc_b1 = (const float*)d_in[5];
  const float* enc_w2 = (const float*)d_in[6];  const float* enc_b2 = (const float*)d_in[7];
  const float* gnn_w1 = (const float*)d_in[8];  const float* gnn_b1 = (const float*)d_in[9];
  const float* gnn_w2 = (const float*)d_in[10]; const float* gnn_b2 = (const float*)d_in[11];
  const float* lp_w1  = (const float*)d_in[12]; const float* lp_b1  = (const float*)d_in[13];
  const float* lp_w2  = (const float*)d_in[14]; const float* lp_b2  = (const float*)d_in[15];
  const float* hp_w1  = (const float*)d_in[16]; const float* hp_b1  = (const float*)d_in[17];
  const float* hp_w2  = (const float*)d_in[18]; const float* hp_b2  = (const float*)d_in[19];
  const float* v_w1   = (const float*)d_in[20]; const float* v_b1   = (const float*)d_in[21];
  const float* v_w2   = (const float*)d_in[22]; const float* v_b2   = (const float*)d_in[23];

  float* out = (float*)d_out;

  // ---- workspace layout ----
  char* ws = (char*)d_ws;
  size_t off = 0;
  auto walloc = [&](size_t bytes) -> void* {
    void* p = ws + off;
    off += (bytes + 255) & ~(size_t)255;
    return p;
  };
  const size_t NTH = (size_t)NT_NODES * H_DIM;
  unsigned short* Wenc1 = (unsigned short*)walloc((size_t)F_DIM * H_DIM * 2);
  unsigned short* Wenc2 = (unsigned short*)walloc((size_t)H_DIM * H_DIM * 2);
  unsigned short* Wg1   = (unsigned short*)walloc((size_t)H_DIM * H_DIM * 2);
  unsigned short* Wg2   = (unsigned short*)walloc((size_t)H_DIM * H_DIM * 2);
  unsigned short* Wlp   = (unsigned short*)walloc((size_t)H_DIM * H_DIM * 2);
  unsigned short* Whp   = (unsigned short*)walloc((size_t)H_DIM * H_DIM * 2);
  unsigned short* Abf   = (unsigned short*)walloc(NTH * 2);
  unsigned short* Bbf   = (unsigned short*)walloc(NTH * 2);
  float*          Tf    = (float*)walloc(NTH * 4);
  float*          Gf    = (float*)walloc(NTH * 4);
  float*          dinv  = (float*)walloc((size_t)NT_NODES * 4);
  float*          acc   = (float*)walloc(128 * 4);

  const int NTf = (int)NTH;   // 13,508,608 elements (fits int)

  // ---- 0. shuffle+convert weights into WMMA B-fragment order (bf16) ----
  k_shuffle_w<<<cdiv(F_DIM * 8, 256), 256, 0, stream>>>(enc_w1, Wenc1, F_DIM);
  k_shuffle_w<<<cdiv(H_DIM * 8, 256), 256, 0, stream>>>(enc_w2, Wenc2, H_DIM);
  k_shuffle_w<<<cdiv(H_DIM * 8, 256), 256, 0, stream>>>(gnn_w1, Wg1, H_DIM);
  k_shuffle_w<<<cdiv(H_DIM * 8, 256), 256, 0, stream>>>(gnn_w2, Wg2, H_DIM);
  k_shuffle_w<<<cdiv(H_DIM * 8, 256), 256, 0, stream>>>(lp_w1, Wlp, H_DIM);
  k_shuffle_w<<<cdiv(H_DIM * 8, 256), 256, 0, stream>>>(hp_w1, Whp, H_DIM);

  // ---- 1. encoder layer 1 (f32 in, bf16 out), batched layer+head nodes ----
  gemm128_wmma<true, true, true><<<N_LAYER / 32, 256, 0, stream>>>(
      (const void*)x, Wenc1, enc_b1, (void*)Abf, F_DIM);
  gemm128_wmma<true, true, true><<<(L_HEADS * NH_NODES) / 32, 256, 0, stream>>>(
      (const void*)head_x, Wenc1, enc_b1,
      (void*)(Abf + (size_t)N_LAYER * H_DIM), F_DIM);

  // ---- 2. encoder layer 2 ----
  gemm128_wmma<false, true, true><<<NT_NODES / 32, 256, 0, stream>>>(
      (const void*)Abf, Wenc2, enc_b2, (void*)Bbf, H_DIM);

  // ---- 3. degree / dinv (same for both GCN layers) ----
  k_fill1<<<cdiv(NT_NODES, 256), 256, 0, stream>>>(dinv, NT_NODES);
  k_deg_layer<<<cdiv(E_LAYER, 256), 256, 0, stream>>>(ei, dinv);
  k_deg_head<<<cdiv(L_HEADS * EH_EDGES, 256), 256, 0, stream>>>(hei, dinv);
  k_finalize_dinv<<<cdiv(NT_NODES, 256), 256, 0, stream>>>(dinv, NT_NODES);

  // ---- 4. GCN layer 1: T = h @ W ; aggregate ; bias+relu -> Abf ----
  gemm128_wmma<false, false, false><<<NT_NODES / 32, 256, 0, stream>>>(
      (const void*)Bbf, Wg1, nullptr, (void*)Tf, H_DIM);
  k_self_loop<<<cdiv(NTf, 256), 256, 0, stream>>>(Tf, dinv, Gf, NTf);
  k_edge_agg_layer<<<cdiv((long long)E_LAYER * 32, 256), 256, 0, stream>>>(ei, Tf, dinv, Gf);
  k_edge_agg_head<<<cdiv((long long)L_HEADS * EH_EDGES * 32, 256), 256, 0, stream>>>(hei, Tf, dinv, Gf);
  k_bias_relu_cvt<false><<<cdiv(NTf, 256), 256, 0, stream>>>(Gf, gnn_b1, Abf, NTf);

  // ---- 5. GCN layer 2 -> final features (bf16 in Bbf, f32 in Gf) ----
  gemm128_wmma<false, false, false><<<NT_NODES / 32, 256, 0, stream>>>(
      (const void*)Abf, Wg2, nullptr, (void*)Tf, H_DIM);
  k_self_loop<<<cdiv(NTf, 256), 256, 0, stream>>>(Tf, dinv, Gf, NTf);
  k_edge_agg_layer<<<cdiv((long long)E_LAYER * 32, 256), 256, 0, stream>>>(ei, Tf, dinv, Gf);
  k_edge_agg_head<<<cdiv((long long)L_HEADS * EH_EDGES * 32, 256), 256, 0, stream>>>(hei, Tf, dinv, Gf);
  k_bias_relu_cvt<true><<<cdiv(NTf, 256), 256, 0, stream>>>(Gf, gnn_b2, Bbf, NTf);

  // ---- 6. layer policy head: relu(h@lp_w1+b) -> Tf ; sigmoid dot ----
  gemm128_wmma<false, true, false><<<N_LAYER / 32, 256, 0, stream>>>(
      (const void*)Bbf, Wlp, lp_b1, (void*)Tf, H_DIM);
  k_policy_out<<<cdiv((long long)N_LAYER * 32, 256), 256, 0, stream>>>(
      Tf, lp_w2, lp_b2, out, N_LAYER);

  // ---- 7. head policy head over the 65536 head-graph nodes ----
  gemm128_wmma<false, true, false><<<(L_HEADS * NH_NODES) / 32, 256, 0, stream>>>(
      (const void*)(Bbf + (size_t)N_LAYER * H_DIM), Whp, hp_b1,
      (void*)(Tf + (size_t)N_LAYER * H_DIM), H_DIM);
  k_policy_out<<<cdiv((long long)L_HEADS * NH_NODES * 32, 256), 256, 0, stream>>>(
      Tf + (size_t)N_LAYER * H_DIM, hp_w2, hp_b2, out + N_LAYER, L_HEADS * NH_NODES);

  // ---- 8. head mask (layer_probs[:8] > 0.5) ----
  k_mask<<<1, 32, 0, stream>>>(out);

  // ---- 9. value head: mean over layer nodes -> tiny MLP ----
  k_zero<<<1, 128, 0, stream>>>(acc, 128);
  k_row_sum<<<cdiv(N_LAYER, 128), 128, 0, stream>>>(Gf, acc);
  k_value<<<1, 128, 0, stream>>>(acc, v_w1, v_b1, v_w2, v_b2,
                                 out + N_LAYER + L_HEADS * NH_NODES + L_HEADS);
}
